// LinearRNN_5866925326616
// MI455X (gfx1250) — matrix operand order repaired and checked by reading, compiled, not run on hardware
//
#include <hip/hip_runtime.h>

// LinearRNN: out = ( scan_t( x @ W_in^T ; diag(W_h) ) ) @ W_out^T
// B=16 T=4096 D_IN=256 HID=512 D_OUT=256, fp32 in/out.
// f16 WMMA GEMMs (f32 accum), software-pipelined loads; chunked parallel
// diagonal scan with fp32 carries (W_h is diagonal by construction).

typedef _Float16 v16h __attribute__((ext_vector_type(16)));
typedef _Float16 v8h  __attribute__((ext_vector_type(8)));
typedef _Float16 v2h  __attribute__((ext_vector_type(2)));
typedef float    v8f  __attribute__((ext_vector_type(8)));
typedef float    v4f  __attribute__((ext_vector_type(4)));

#define BB    16
#define TT    4096
#define DIN   256
#define HID   512
#define DOUT  256
#define MROWS (BB*TT)        // 65536
#define CHUNK  64
#define NCHUNK (TT/CHUNK)    // 64

static __device__ __forceinline__ v16h combine16(v8h lo, v8h hi) {
    v16h r;
#pragma unroll
    for (int i = 0; i < 8; ++i) { r[i] = lo[i]; r[8 + i] = hi[i]; }
    return r;
}

// A tile fragment (16x32 f16) from an f32 row: elems 0..7 <-> K=k0+8h+0..7,
// elems 8..15 <-> K=k0+16+8h+0..7 (ISA 16-bit A 16x32 layout).
static __device__ __forceinline__ v16h load_a_f32(const float* __restrict__ xrow,
                                                  int k0, int half) {
    const v4f* pa0 = (const v4f*)(xrow + k0 + 8 * half);
    const v4f* pa1 = (const v4f*)(xrow + k0 + 16 + 8 * half);
    v4f f0 = pa0[0], f1 = pa0[1], f2 = pa1[0], f3 = pa1[1];
    v16h a;
#pragma unroll
    for (int i = 0; i < 4; ++i) {
        a[i]      = (_Float16)f0[i];
        a[4 + i]  = (_Float16)f1[i];
        a[8 + i]  = (_Float16)f2[i];
        a[12 + i] = (_Float16)f3[i];
    }
    return a;
}

// A tile fragment from an f16 row (same layout, direct loads).
static __device__ __forceinline__ v16h load_a_f16(const _Float16* __restrict__ arow,
                                                  int k0, int half) {
    v8h a0 = *(const v8h*)(arow + k0 + 8 * half);
    v8h a1 = *(const v8h*)(arow + k0 + 16 + 8 * half);
    return combine16(a0, a1);
}

// B tile fragment (KxN, 16 cols): lane = col n, elems 0..7 <-> K=k0+8h+0..7, elems 8..15 <-> K=k0+16+8h+0..7.
static __device__ __forceinline__ v16h load_b_f16(const _Float16* __restrict__ p) {
    const v8h* pb = (const v8h*)p;
    return combine16(pb[0], pb[2]);
}

// ---- prep: cast weights to f16, extract diag(W_h) --------------------------
__global__ void prep_kernel(const float* __restrict__ Win,
                            const float* __restrict__ Wh,
                            const float* __restrict__ Wout,
                            _Float16* __restrict__ WinH,
                            _Float16* __restrict__ WoutH,
                            float* __restrict__ dg) {
    int i = blockIdx.x * blockDim.x + threadIdx.x;
    const int NW = HID * DIN; // 131072 (== DOUT*HID too)
    if (i < NW) {
        WinH[i] = (_Float16)Win[i];
    } else if (i < 2 * NW) {
        WoutH[i - NW] = (_Float16)Wout[i - NW];
    } else if (i < 2 * NW + HID) {
        int j = i - 2 * NW;
        dg[j] = Wh[(size_t)j * HID + j];   // diagonal of W_h
    }
}

// ---- GEMM1: xp[M,512] = x[M,256] @ W_in^T, store f16 -----------------------
// grid.x = M/16 (4096); 8 waves/block each own a 16x64 slice of N (full N=512).
// 2-stage software pipeline: K-step k+1 loads issue before K-step k WMMAs retire.
__global__ __launch_bounds__(256)
void gemm_in_kernel(const float* __restrict__ x,
                    const _Float16* __restrict__ WinH,
                    _Float16* __restrict__ xp) {
    const int lane = threadIdx.x & 31;
    const int wave = threadIdx.x >> 5;          // 0..7
    const int half = lane >> 4;                 // lane group
    const int l16  = lane & 15;
    const int m    = blockIdx.x * 16 + l16;     // A row owned by this lane
    const int n0   = wave * 64;

    v8f acc[4] = {};
    const float* xrow = x + (size_t)m * DIN;
    const _Float16* brow[4];
#pragma unroll
    for (int s = 0; s < 4; ++s)
        brow[s] = WinH + (size_t)(n0 + s * 16 + l16) * DIN + 8 * half;

    v16h a = load_a_f32(xrow, 0, half);
    v16h b[4];
#pragma unroll
    for (int s = 0; s < 4; ++s) b[s] = load_b_f16(brow[s]);

#pragma unroll
    for (int k0 = 0; k0 < DIN; k0 += 32) {
        v16h an, bn[4];
        if (k0 + 32 < DIN) {                    // compile-time after unroll
            an = load_a_f32(xrow, k0 + 32, half);
#pragma unroll
            for (int s = 0; s < 4; ++s) bn[s] = load_b_f16(brow[s] + k0 + 32);
        }
#pragma unroll
        for (int s = 0; s < 4; ++s)
            acc[s] = __builtin_amdgcn_wmma_f32_16x16x32_f16(
                false, a, false, b[s], (short)0, acc[s], false, false);
        if (k0 + 32 < DIN) {
            a = an;
#pragma unroll
            for (int s = 0; s < 4; ++s) b[s] = bn[s];
        }
    }
    // C/D: elem r <-> row 8h+r, lane <-> col
#pragma unroll
    for (int s = 0; s < 4; ++s) {
        int n = n0 + s * 16 + l16;
#pragma unroll
        for (int r = 0; r < 8; ++r) {
            int mm = blockIdx.x * 16 + 8 * half + r;
            xp[(size_t)mm * HID + n] = (_Float16)acc[s][r];
        }
    }
}

// ---- scan phase A: per-chunk local scan, in place, carry-in = 0 ------------
// grid = B*NCHUNK blocks, 256 threads, 2 channels/thread.
__global__ __launch_bounds__(256)
void scan_local_kernel(_Float16* __restrict__ xp,
                       const float* __restrict__ dg,
                       float* __restrict__ carry) {
    int b = blockIdx.x >> 6;            // / NCHUNK
    int c = blockIdx.x & (NCHUNK - 1);
    int j = threadIdx.x * 2;
    float d0 = dg[j], d1 = dg[j + 1];
    float h0 = 0.f, h1 = 0.f;
    _Float16* p = xp + ((size_t)(b * TT + c * CHUNK) * HID) + j;
#pragma unroll 4
    for (int t = 0; t < CHUNK; ++t) {
        v2h v = *(v2h*)p;
        h0 = (float)v[0] + d0 * h0;
        h1 = (float)v[1] + d1 * h1;
        v[0] = (_Float16)h0;
        v[1] = (_Float16)h1;
        *(v2h*)p = v;
        p += HID;
    }
    float* cb = carry + ((size_t)(b * NCHUNK + c) * HID) + j;
    cb[0] = h0;
    cb[1] = h1;
}

// ---- scan phase B: cross-chunk prefix: H_c = E_c + d^CHUNK * H_{c-1} -------
// Overwrites carry[b][c][j] with the INCOMING carry for chunk c (= H_{c-1}).
__global__ void scan_prefix_kernel(float* __restrict__ carry,
                                   const float* __restrict__ dg) {
    int b = blockIdx.x;
    int j = threadIdx.x;                 // 512 threads
    float dj = dg[j];
    float dC = dj;                       // d^64 via 6 squarings
#pragma unroll
    for (int i = 0; i < 6; ++i) dC *= dC;
    float H = 0.f;
    for (int c = 0; c < NCHUNK; ++c) {
        float* e = carry + ((size_t)(b * NCHUNK + c) * HID) + j;
        float E = *e;
        *e = H;                          // incoming carry for chunk c
        H = E + dC * H;
    }
}

// ---- scan phase C: apply carry: h_t += d^{t+1} * H_in ----------------------
__global__ __launch_bounds__(256)
void scan_fix_kernel(_Float16* __restrict__ xp,
                     const float* __restrict__ dg,
                     const float* __restrict__ carry) {
    int b = blockIdx.x >> 6;
    int c = blockIdx.x & (NCHUNK - 1);
    if (c == 0) return;                  // carry-in is zero
    int j = threadIdx.x * 2;
    float d0 = dg[j], d1 = dg[j + 1];
    const float* cb = carry + ((size_t)(b * NCHUNK + c) * HID) + j;
    float I0 = cb[0], I1 = cb[1];
    float p0 = d0, p1 = d1;
    _Float16* p = xp + ((size_t)(b * TT + c * CHUNK) * HID) + j;
#pragma unroll 4
    for (int t = 0; t < CHUNK; ++t) {
        v2h v = *(v2h*)p;
        v[0] = (_Float16)((float)v[0] + p0 * I0);
        v[1] = (_Float16)((float)v[1] + p1 * I1);
        *(v2h*)p = v;
        p0 *= d0;
        p1 *= d1;
        p += HID;
    }
}

// ---- GEMM2: out[M,256] = hs[M,512] @ W_out^T, f32 out ----------------------
// grid.x = M/32 (2048); 8 waves: waves 0-3 -> M tile 0, waves 4-7 -> M tile 1.
// Same 2-stage pipeline as GEMM1.
__global__ __launch_bounds__(256)
void gemm_out_kernel(const _Float16* __restrict__ hs,
                     const _Float16* __restrict__ WoutH,
                     float* __restrict__ out) {
    const int lane = threadIdx.x & 31;
    const int wave = threadIdx.x >> 5;
    const int half = lane >> 4;
    const int l16  = lane & 15;
    const int tile_m = blockIdx.x * 2 + (wave >> 2);
    const int n0     = (wave & 3) * 64;
    const int m      = tile_m * 16 + l16;

    v8f acc[4] = {};
    const _Float16* arow = hs + (size_t)m * HID;
    const _Float16* brow[4];
#pragma unroll
    for (int s = 0; s < 4; ++s)
        brow[s] = WoutH + (size_t)(n0 + s * 16 + l16) * HID + 8 * half;

    v16h a = load_a_f16(arow, 0, half);
    v16h b[4];
#pragma unroll
    for (int s = 0; s < 4; ++s) b[s] = load_b_f16(brow[s]);

#pragma unroll
    for (int k0 = 0; k0 < HID; k0 += 32) {
        v16h an, bn[4];
        if (k0 + 32 < HID) {
            an = load_a_f16(arow, k0 + 32, half);
#pragma unroll
            for (int s = 0; s < 4; ++s) bn[s] = load_b_f16(brow[s] + k0 + 32);
        }
#pragma unroll
        for (int s = 0; s < 4; ++s)
            acc[s] = __builtin_amdgcn_wmma_f32_16x16x32_f16(
                false, a, false, b[s], (short)0, acc[s], false, false);
        if (k0 + 32 < HID) {
            a = an;
#pragma unroll
            for (int s = 0; s < 4; ++s) b[s] = bn[s];
        }
    }
#pragma unroll
    for (int s = 0; s < 4; ++s) {
        int n = n0 + s * 16 + l16;
#pragma unroll
        for (int r = 0; r < 8; ++r) {
            int mm = tile_m * 16 + 8 * half + r;
            out[(size_t)mm * DOUT + n] = acc[s][r];
        }
    }
}

extern "C" void kernel_launch(void* const* d_in, const int* in_sizes, int n_in,
                              void* d_out, int out_size, void* d_ws, size_t ws_size,
                              hipStream_t stream) {
    const float* x    = (const float*)d_in[0]; // [16,4096,256]
    const float* Win  = (const float*)d_in[1]; // [512,256]
    const float* Wh   = (const float*)d_in[2]; // [512,512] (diagonal)
    const float* Wout = (const float*)d_in[3]; // [256,512]
    float* out = (float*)d_out;                // [16,4096,256]

    // workspace layout (~66.5 MB total)
    char* ws = (char*)d_ws;
    size_t off = 0;
    _Float16* xp    = (_Float16*)(ws + off); off += (size_t)MROWS * HID * 2;   // 64 MB (xp -> hs in place)
    _Float16* WinH  = (_Float16*)(ws + off); off += (size_t)HID * DIN * 2;     // 256 KB
    _Float16* WoutH = (_Float16*)(ws + off); off += (size_t)DOUT * HID * 2;    // 256 KB
    float*    dg    = (float*)   (ws + off); off += (size_t)HID * 4;           // 2 KB
    float*    carry = (float*)   (ws + off); off += (size_t)BB * NCHUNK * HID * 4; // 2 MB

    const int prepN = 2 * HID * DIN + HID;
    prep_kernel<<<(prepN + 255) / 256, 256, 0, stream>>>(Win, Wh, Wout, WinH, WoutH, dg);
    gemm_in_kernel<<<MROWS / 16, 256, 0, stream>>>(x, WinH, xp);
    scan_local_kernel<<<BB * NCHUNK, 256, 0, stream>>>(xp, dg, carry);
    scan_prefix_kernel<<<BB, HID, 0, stream>>>(carry, dg);
    scan_fix_kernel<<<BB * NCHUNK, 256, 0, stream>>>(xp, dg, carry);
    gemm_out_kernel<<<MROWS / 32, 256, 0, stream>>>(xp, WoutH, out);
}
